// Attention_69544110457345
// MI455X (gfx1250) — matrix operand order repaired and checked by reading, not compiled, not run on hardware
//
#include <hip/hip_runtime.h>
#include <cstddef>
#include <cstdint>

// ---------------------------------------------------------------------------
// Types for CDNA5 WMMA (wave32, 16x16x32 bf16 -> f32)
// ---------------------------------------------------------------------------
typedef __bf16 bf16_t;
typedef __attribute__((ext_vector_type(16))) __bf16 v16bf;
typedef __attribute__((ext_vector_type(8)))  __bf16 v8bf;
typedef __attribute__((ext_vector_type(8)))  float  v8f;
typedef __attribute__((ext_vector_type(4)))  unsigned int v4u;
typedef __attribute__((ext_vector_type(8)))  int v8i;
typedef __attribute__((ext_vector_type(4)))  int v4i;

#if defined(__AMDGCN__) && __has_builtin(__builtin_amdgcn_tensor_load_to_lds) && \
    __has_builtin(__builtin_amdgcn_s_wait_tensorcnt)
#define USE_TDM 1
#endif

__device__ __forceinline__ v8f v8f_zero() {
    v8f z;
#pragma unroll
    for (int i = 0; i < 8; ++i) z[i] = 0.0f;
    return z;
}

__device__ __forceinline__ v16bf join8(v8bf lo, v8bf hi) {
    v16bf r;
#pragma unroll
    for (int i = 0; i < 8; ++i) { r[i] = lo[i]; r[i + 8] = hi[i]; }
    return r;
}

__device__ __forceinline__ v8f wmma_bf16(v16bf a, v16bf b, v8f c) {
    return __builtin_amdgcn_wmma_f32_16x16x32_bf16(
        /*neg_a=*/false, a, /*neg_b=*/false, b,
        /*c_mod=*/(short)0, c, /*reuse_a=*/false, /*reuse_b=*/false);
}

// A fragment (16x32, row-major source): lane<16 row=l holds K {0..7,16..23},
// lane>=16 row=l holds K {8..15,24..31}  (ISA 16-bit A layout).
__device__ __forceinline__ v16bf load_a_frag(const bf16_t* __restrict__ A,
                                             size_t ld, int row, int kbase, int hf) {
    const bf16_t* p = A + (size_t)row * ld + kbase + hf * 8;
    return join8(*(const v8bf*)p, *(const v8bf*)(p + 16));
}

#ifdef USE_TDM
// Issue a TDM DMA of a 2D bf16 tile: 32 rows x 128 elements, row stride 1024
// elements, from global memory into LDS at byte offset lds_off.
// D# layout per CDNA5 ISA (group0 128b, group1 256b).
__device__ __forceinline__ void tdm_load_32x128_bf16(const void* gaddr,
                                                     unsigned int lds_off) {
    unsigned long long ga = (unsigned long long)(uintptr_t)gaddr;
    v4u g0;
    g0[0] = 1u;                                           // count=1, no gather
    g0[1] = lds_off;                                      // lds_addr (bytes)
    g0[2] = (unsigned int)(ga & 0xFFFFFFFFull);           // global_addr[95:64]
    g0[3] = (unsigned int)((ga >> 32) & 0x1FFFFFFull)     // global_addr[120:96]
          | (2u << 30);                                   // type=2 ("image")
    v8i g1;
    g1[0] = 0x00010000;            // workgroup_mask=0, data_size=1 (2 bytes)
    g1[1] = (int)(128u << 16);     // tensor_dim0 = 128  (bits 79:48, low half)
    g1[2] = (int)(32u << 16);      // tensor_dim1 = 32   (bits 111:80, low half)
    g1[3] = (int)(128u << 16);     // tile_dim0  = 128   (bits 127:112)
    g1[4] = 32;                    // tile_dim1 = 32, tile_dim2 = 0
    g1[5] = 1024;                  // tensor_dim0_stride = 1024 elements
    g1[6] = 0;                     // stride0 hi, stride1 lo
    g1[7] = 0;                     // stride1 hi
    v4i z4; z4[0] = z4[1] = z4[2] = z4[3] = 0;
#if __clang_major__ >= 23
    v8i z8;
#pragma unroll
    for (int i = 0; i < 8; ++i) z8[i] = 0;
    __builtin_amdgcn_tensor_load_to_lds(g0, g1, z4, z4, z8, 0);
#else
    __builtin_amdgcn_tensor_load_to_lds(g0, g1, z4, z4, 0);
#endif
}
#endif

// ---------------------------------------------------------------------------
// Elementwise casts
// ---------------------------------------------------------------------------
__global__ void cast_f32_bf16(const float* __restrict__ in, bf16_t* __restrict__ out,
                              size_t n) {
    size_t i = (size_t)blockIdx.x * blockDim.x + threadIdx.x;
    if (i < n) out[i] = (bf16_t)in[i];
}

// in: f32 [K, N] row-major  ->  out: bf16 [N, K] row-major (i.e. W^T)
__global__ __launch_bounds__(256) void transpose_cast(const float* __restrict__ in,
                                                      bf16_t* __restrict__ out,
                                                      int K, int N) {
    __shared__ float tile[32][33];
    int bx = blockIdx.x * 32;   // N direction
    int by = blockIdx.y * 32;   // K direction
    int tx = threadIdx.x, ty = threadIdx.y;
#pragma unroll
    for (int j = ty; j < 32; j += 8)
        tile[j][tx] = in[(size_t)(by + j) * N + (bx + tx)];
    __syncthreads();
#pragma unroll
    for (int j = ty; j < 32; j += 8)
        out[(size_t)(bx + j) * K + (by + tx)] = (bf16_t)tile[tx][j];
}

// ---------------------------------------------------------------------------
// GEMM: C[M,N] = A[M,K](bf16,row-major) * B^T[N,K](bf16,row-major)
// One wave per 64x64 tile, 4x4 WMMA register blocking, K-step 32.
// 16 WMMAs per 16 b128 fragment loads -> 2x intensity of 2x2 blocking.
// ---------------------------------------------------------------------------
template <bool BF16OUT>
__global__ __launch_bounds__(32) void gemm_wmma(const bf16_t* __restrict__ A,
                                                const bf16_t* __restrict__ BT,
                                                void* __restrict__ C,
                                                int M, int N, int K) {
    const int col0 = blockIdx.x * 64;
    const int row0 = blockIdx.y * 64;
    const int lane = threadIdx.x;
    const int l = lane & 15, hf = lane >> 4;

    v8f acc[4][4];
#pragma unroll
    for (int mt = 0; mt < 4; ++mt)
#pragma unroll
        for (int nt = 0; nt < 4; ++nt) acc[mt][nt] = v8f_zero();

    for (int k0 = 0; k0 < K; k0 += 32) {
        v16bf a[4], b[4];
#pragma unroll
        for (int mt = 0; mt < 4; ++mt)
            a[mt] = load_a_frag(A, K, row0 + mt * 16 + l, k0, hf);
#pragma unroll
        for (int nt = 0; nt < 4; ++nt)
            b[nt] = join8(*(const v8bf*)(BT + (size_t)(col0 + nt * 16 + l) * K + k0 + hf * 8),
                          *(const v8bf*)(BT + (size_t)(col0 + nt * 16 + l) * K + k0 + hf * 8 + 16));
#pragma unroll
        for (int mt = 0; mt < 4; ++mt)
#pragma unroll
            for (int nt = 0; nt < 4; ++nt)
                acc[mt][nt] = wmma_bf16(a[mt], b[nt], acc[mt][nt]);
    }

#pragma unroll
    for (int mt = 0; mt < 4; ++mt)
#pragma unroll
        for (int nt = 0; nt < 4; ++nt)
#pragma unroll
            for (int j = 0; j < 8; ++j) {
                int row = row0 + mt * 16 + j + hf * 8;  // C layout: VGPR j, half-wave
                int col = col0 + nt * 16 + l;
                float v = acc[mt][nt][j];
                if (BF16OUT) ((bf16_t*)C)[(size_t)row * N + col] = (bf16_t)v;
                else         ((float*)C)[(size_t)row * N + col]  = v;
            }
}

// ---------------------------------------------------------------------------
// RoPE, in place on bf16 tensor [tokens, nheads, 128], interleaved pairs
// ---------------------------------------------------------------------------
__global__ void rope_bf16(bf16_t* __restrict__ t, const float* __restrict__ cosT,
                          const float* __restrict__ sinT, int nheads, int total) {
    int idx = blockIdx.x * blockDim.x + threadIdx.x;
    if (idx >= total) return;
    int i    = idx & 63;            // pair index 0..63
    int rest = idx >> 6;
    int hh   = rest % nheads;
    int tok  = rest / nheads;
    int s    = tok & 2047;          // SEQ = 2048
    float c  = cosT[s * 64 + i];
    float sn = sinT[s * 64 + i];
    size_t base = (size_t)tok * nheads * 128 + (size_t)hh * 128 + 2 * i;
    float a = (float)t[base], b = (float)t[base + 1];
    t[base]     = (bf16_t)(a * c - b * sn);
    t[base + 1] = (bf16_t)(a * sn + b * c);
}

// ---------------------------------------------------------------------------
// Flash attention: one wave per (b, head, 16-row q block).
// K and V 32x128 blocks are staged into LDS by the Tensor Data Mover
// (tensor_load_to_lds + s_wait_tensorcnt), WMMAs read fragments from LDS.
// ---------------------------------------------------------------------------
__global__ __launch_bounds__(32) void flash_attn(const bf16_t* __restrict__ qb,
                                                 const bf16_t* __restrict__ kb,
                                                 const bf16_t* __restrict__ vb,
                                                 bf16_t* __restrict__ ob) {
    const int S = 2048, HD = 128, QD = 4096, KVD = 1024;
    const int qi0  = blockIdx.x * 16;
    const int h    = blockIdx.y;
    const int bi   = blockIdx.z;
    const int kvh  = h >> 2;                     // N_REP = 4
    const int lane = threadIdx.x;
    const int l = lane & 15, hf = lane >> 4;
    const float scale = 0.08838834764831845f;    // 1/sqrt(128)

    __shared__ alignas(32) bf16_t kl[32 * 128];  // staged K block (8 KB)
    __shared__ alignas(32) bf16_t vl[32 * 128];  // staged V block (8 KB)
    __shared__ alignas(16) bf16_t pl[16 * 32];   // P tile layout transpose (1 KB)

    // Q fragments for the whole 16x128 block (4 x K=32 chunks)
    v16bf qf[4];
#pragma unroll
    for (int kc = 0; kc < 4; ++kc) {
        const bf16_t* p = qb + (size_t)(bi * S + qi0 + l) * QD + h * HD + kc * 32 + hf * 8;
        qf[kc] = join8(*(const v8bf*)p, *(const v8bf*)(p + 16));
    }

    float m[8], lsum[8];
    v8f o[8];
#pragma unroll
    for (int j = 0; j < 8; ++j) { m[j] = -1.0e30f; lsum[j] = 0.0f; }
#pragma unroll
    for (int t = 0; t < 8; ++t) o[t] = v8f_zero();

    for (int j0 = 0; j0 < qi0 + 16; j0 += 32) {
        // ---- stage K and V blocks (32 keys x 128 dims) into LDS ----
        const bf16_t* kg = kb + (size_t)(bi * S + j0) * KVD + kvh * HD;
        const bf16_t* vg = vb + (size_t)(bi * S + j0) * KVD + kvh * HD;
#ifdef USE_TDM
        tdm_load_32x128_bf16(kg, (unsigned int)(uintptr_t)&kl[0]);
        tdm_load_32x128_bf16(vg, (unsigned int)(uintptr_t)&vl[0]);
        __builtin_amdgcn_s_wait_tensorcnt(0);
#else
        {
            const v16bf* ks = (const v16bf*)(kg + (size_t)lane * KVD);
            const v16bf* vs = (const v16bf*)(vg + (size_t)lane * KVD);
            v16bf* kd = (v16bf*)(kl + lane * HD);
            v16bf* vd = (v16bf*)(vl + lane * HD);
#pragma unroll
            for (int i = 0; i < 8; ++i) { kd[i] = ks[i]; vd[i] = vs[i]; }
        }
        __syncthreads();
#endif

        // ---- S = Q * K^T for 32 keys (two 16x16 tiles) from LDS ----
        v8f s0 = v8f_zero(), s1 = v8f_zero();
#pragma unroll
        for (int kc = 0; kc < 4; ++kc) {
            // B frag: lane column = key l (+16 for tile 1); dims {8hf..8hf+7, 16+8hf..16+8hf+7}
            v16bf b0 = join8(*(const v8bf*)(kl + (l)      * HD + kc * 32 + hf * 8),
                             *(const v8bf*)(kl + (l)      * HD + kc * 32 + hf * 8 + 16));
            v16bf b1 = join8(*(const v8bf*)(kl + (16 + l) * HD + kc * 32 + hf * 8),
                             *(const v8bf*)(kl + (16 + l) * HD + kc * 32 + hf * 8 + 16));
            s0 = wmma_bf16(qf[kc], b0, s0);
            s1 = wmma_bf16(qf[kc], b1, s1);
        }

        // ---- online softmax (row stats via width-16 shfl reductions) ----
#pragma unroll
        for (int j = 0; j < 8; ++j) {
            int row = qi0 + j + hf * 8;
            float a0 = s0[j] * scale + ((j0 + l      > row) ? -1.0e9f : 0.0f);
            float a1 = s1[j] * scale + ((j0 + 16 + l > row) ? -1.0e9f : 0.0f);
            float rmax = fmaxf(a0, a1);
#pragma unroll
            for (int off = 8; off; off >>= 1) rmax = fmaxf(rmax, __shfl_xor(rmax, off, 16));
            float mn    = fmaxf(m[j], rmax);
            float alpha = __expf(m[j] - mn);
            float p0 = __expf(a0 - mn);
            float p1 = __expf(a1 - mn);
            float rs = p0 + p1;
#pragma unroll
            for (int off = 8; off; off >>= 1) rs += __shfl_xor(rs, off, 16);
            lsum[j] = lsum[j] * alpha + rs;
            m[j] = mn;
#pragma unroll
            for (int t = 0; t < 8; ++t) o[t][j] *= alpha;   // rescale accumulator
            pl[(j + hf * 8) * 32 + l]      = (bf16_t)p0;    // C-layout -> LDS row-major
            pl[(j + hf * 8) * 32 + 16 + l] = (bf16_t)p1;
        }
        __syncthreads();

        // ---- P fragment (A layout, 16x32) from LDS ----
        const bf16_t* pp = pl + l * 32 + hf * 8;
        v16bf pf = join8(*(const v8bf*)pp, *(const v8bf*)(pp + 16));

        // ---- O += P * V ----
#pragma unroll
        for (int t2 = 0; t2 < 8; ++t2) {
            v16bf vf;
#pragma unroll
            for (int e = 0; e < 16; ++e) vf[e] = vl[(hf * 8 + (e & 7) + ((e >> 3) << 4)) * HD + t2 * 16 + l];
            o[t2] = wmma_bf16(pf, vf, o[t2]);
        }
        __syncthreads();
    }

    // ---- normalize + store bf16 attention output ----
#pragma unroll
    for (int j = 0; j < 8; ++j) {
        float inv = 1.0f / lsum[j];
        int row = qi0 + j + hf * 8;
        bf16_t* outp = ob + (size_t)(bi * S + row) * QD + h * HD;
#pragma unroll
        for (int t2 = 0; t2 < 8; ++t2)
            outp[t2 * 16 + l] = (bf16_t)(o[t2][j] * inv);
    }
}

// ---------------------------------------------------------------------------
// Host launcher
// ---------------------------------------------------------------------------
extern "C" void kernel_launch(void* const* d_in, const int* in_sizes, int n_in,
                              void* d_out, int out_size, void* d_ws, size_t ws_size,
                              hipStream_t stream) {
    (void)in_sizes; (void)n_in; (void)out_size; (void)ws_size;
    const float* x    = (const float*)d_in[0];
    const float* wq   = (const float*)d_in[1];
    const float* wk   = (const float*)d_in[2];
    const float* wv   = (const float*)d_in[3];
    const float* wo   = (const float*)d_in[4];
    const float* fcos = (const float*)d_in[7];
    const float* fsin = (const float*)d_in[8];
    float* out = (float*)d_out;

    const int BSZ = 2, SEQ = 2048, DIM = 4096, KVDIM = 1024;
    const int TOK = BSZ * SEQ;                    // 4096 token rows

    char* ws = (char*)d_ws;
    size_t off = 0;
    auto alloc = [&](size_t bytes) { void* p = ws + off; off += bytes; return p; };
    bf16_t* xb    = (bf16_t*)alloc((size_t)TOK * DIM   * 2);
    bf16_t* wqT   = (bf16_t*)alloc((size_t)DIM * DIM   * 2);
    bf16_t* wkT   = (bf16_t*)alloc((size_t)KVDIM * DIM * 2);
    bf16_t* wvT   = (bf16_t*)alloc((size_t)KVDIM * DIM * 2);
    bf16_t* woT   = (bf16_t*)alloc((size_t)DIM * DIM   * 2);
    bf16_t* qb    = (bf16_t*)alloc((size_t)TOK * DIM   * 2);
    bf16_t* kbuf  = (bf16_t*)alloc((size_t)TOK * KVDIM * 2);
    bf16_t* vbuf  = (bf16_t*)alloc((size_t)TOK * KVDIM * 2);
    bf16_t* attnb = (bf16_t*)alloc((size_t)TOK * DIM   * 2);

    // 1) casts / weight transposes
    {
        size_t n = (size_t)TOK * DIM;
        cast_f32_bf16<<<(unsigned)((n + 255) / 256), 256, 0, stream>>>(x, xb, n);
    }
    transpose_cast<<<dim3(DIM / 32,   DIM / 32), dim3(32, 8), 0, stream>>>(wq, wqT, DIM, DIM);
    transpose_cast<<<dim3(KVDIM / 32, DIM / 32), dim3(32, 8), 0, stream>>>(wk, wkT, DIM, KVDIM);
    transpose_cast<<<dim3(KVDIM / 32, DIM / 32), dim3(32, 8), 0, stream>>>(wv, wvT, DIM, KVDIM);
    transpose_cast<<<dim3(DIM / 32,   DIM / 32), dim3(32, 8), 0, stream>>>(wo, woT, DIM, DIM);

    // 2) QKV projections (WMMA, bf16 out), one wave per 64x64 tile
    gemm_wmma<true><<<dim3(DIM / 64,   TOK / 64), 32, 0, stream>>>(xb, wqT, qb,   TOK, DIM,   DIM);
    gemm_wmma<true><<<dim3(KVDIM / 64, TOK / 64), 32, 0, stream>>>(xb, wkT, kbuf, TOK, KVDIM, DIM);
    gemm_wmma<true><<<dim3(KVDIM / 64, TOK / 64), 32, 0, stream>>>(xb, wvT, vbuf, TOK, KVDIM, DIM);

    // 3) RoPE (start_pos == 0; cache update is a no-op for the returned output)
    {
        int totq = TOK * 32 * 64;
        rope_bf16<<<(totq + 255) / 256, 256, 0, stream>>>(qb, fcos, fsin, 32, totq);
        int totk = TOK * 8 * 64;
        rope_bf16<<<(totk + 255) / 256, 256, 0, stream>>>(kbuf, fcos, fsin, 8, totk);
    }

    // 4) flash attention (causal) with TDM-staged K/V blocks
    flash_attn<<<dim3(SEQ / 16, 32, BSZ), 32, 0, stream>>>(qb, kbuf, vbuf, attnb);

    // 5) output projection (WMMA, f32 out)
    gemm_wmma<false><<<dim3(DIM / 64, TOK / 64), 32, 0, stream>>>(attnb, woT, out, TOK, DIM, DIM);
}
